// LearnablePhi_27633819583147
// MI455X (gfx1250) — matrix operand order repaired and checked by reading, not compiled, not run on hardware
//
#include <hip/hip_runtime.h>
#include <math.h>

#define HID 128
#define EMB 64
#define BB  128
#define NN  4096
#define KPAIR 32
#define ROWS 64
#define APITCH 136   // 128 + 8 halves pad (16B-aligned rows, avoids bank conflicts)
#define PPITCH 72    // 64 + 8 halves pad for pair kernel

typedef __attribute__((ext_vector_type(16))) _Float16 v16h;
typedef __attribute__((ext_vector_type(8)))  _Float16 v8h;
typedef __attribute__((ext_vector_type(8)))  float    v8f;

__device__ __forceinline__ float geluf(float x) {
  return 0.5f * x * (1.0f + erff(x * 0.7071067811865476f));
}

__device__ __forceinline__ void atomicMaxF(float* a, float v) {
  int old = __float_as_int(*a);
  while (v > __int_as_float(old)) {
    int prev = atomicCAS((int*)a, old, __float_as_int(v));
    if (prev == old) break;
    old = prev;
  }
}

__device__ __forceinline__ v8f wmma16(v16h a, v16h b, v8f c) {
  // v_wmma_f32_16x16x32_f16  D = A*B + C  (f32 accumulate)
  return __builtin_amdgcn_wmma_f32_16x16x32_f16(false, a, false, b, (short)0, c,
                                                false, false);
}

// A fragment (16x32, f16) from a row-major f32 source row.
// Per ISA: lanes 0-15 (row=lane) hold K = {k0..k0+7, k0+16..k0+23};
// lanes 16-31 hold the same pattern shifted by +8 (k0 already includes +8).
__device__ __forceinline__ v16h a_frag_f32(const float* __restrict__ row, int k0) {
  float t[16];
  const float4* p0 = (const float4*)(row + k0);
  const float4* p1 = (const float4*)(row + k0 + 16);
  *(float4*)&t[0]  = p0[0];
  *(float4*)&t[4]  = p0[1];
  *(float4*)&t[8]  = p1[0];
  *(float4*)&t[12] = p1[1];
  v16h a;
#pragma unroll
  for (int i = 0; i < 16; ++i) a[i] = (_Float16)t[i];
  return a;
}

// A fragment from an f16 LDS row (two 16B ds_load_b128).
__device__ __forceinline__ v16h a_frag_lds(const _Float16* row, int k0) {
  v8h lo = *(const v8h*)(row + k0);
  v8h hi = *(const v8h*)(row + k0 + 16);
  v16h a;
#pragma unroll
  for (int i = 0; i < 8; ++i) { a[i] = lo[i]; a[i + 8] = hi[i]; }
  return a;
}

// B fragment from pre-packed weights: fragment (nt,kt) stored as 32 lanes x 16 halves.
__device__ __forceinline__ v16h b_frag(const _Float16* __restrict__ wp, int nt,
                                       int kt, int KT, int lane) {
  const v8h* p = (const v8h*)(wp + ((size_t)(nt * KT + kt) * 32 + lane) * 16);
  v8h lo = p[0];
  v8h hi = p[1];
  v16h b;
#pragma unroll
  for (int i = 0; i < 8; ++i) { b[i] = lo[i]; b[i + 8] = hi[i]; }
  return b;
}

// Pack f32 weight W[K][N] (row-major) into B-fragment layout:
// idx = ((nt*KT + kt)*32 + lane)*16 + hh ;  n = nt*16 + lane%16 ;
// k = kt*32 + (hh&7) + (hh>=8 ? 16 : 0) + (lane>=16 ? 8 : 0)
__global__ void pack_kernel(const float* __restrict__ W, _Float16* __restrict__ out,
                            int Kdim, int Ndim) {
  int idx = blockIdx.x * 256 + threadIdx.x;
  if (idx >= Kdim * Ndim) return;
  int hh   = idx & 15;
  int lane = (idx >> 4) & 31;
  int frag = idx >> 9;
  int KT = Kdim >> 5;
  int nt = frag / KT;
  int kt = frag - nt * KT;
  int n = nt * 16 + (lane & 15);
  int k = kt * 32 + (hh & 7) + ((hh >> 3) << 4) + ((lane >= 16) ? 8 : 0);
  out[idx] = (_Float16)W[k * Ndim + n];
}

__global__ void init_kernel(float* __restrict__ pool_sum, float* __restrict__ pool_max) {
  int i = blockIdx.x * 256 + threadIdx.x;
  if (i < BB * EMB) {
    pool_sum[i] = 0.0f;
    pool_max[i] = -3.4028235e38f;
  }
}

__global__ void nvals_kernel(const unsigned char* __restrict__ mask,
                             float* __restrict__ nvals) {
  __shared__ float sh[256];
  int b = blockIdx.x;
  float s = 0.0f;
  for (int i = threadIdx.x; i < NN; i += 256)
    s += mask[(size_t)b * NN + i] ? 1.0f : 0.0f;
  sh[threadIdx.x] = s;
  __syncthreads();
  for (int o = 128; o > 0; o >>= 1) {
    if (threadIdx.x < o) sh[threadIdx.x] += sh[threadIdx.x + o];
    __syncthreads();
  }
  if (threadIdx.x == 0) nvals[b] = sh[0];
}

// Cell encoder: 3-layer MLP with LN+GELU, WMMA-tiled.
// mode 0: rows = blockIdx*64..  -> masked pooling via atomics
// mode 1: rows gathered through `gather` (pair_idx flat) -> ce_out [rows x 64]
__global__ __launch_bounds__(128) void ce_kernel(
    const float* __restrict__ x, const unsigned char* __restrict__ mask,
    const int* __restrict__ gather,
    const _Float16* __restrict__ wp1, const float* __restrict__ b1,
    const float* __restrict__ g1, const float* __restrict__ be1,
    const _Float16* __restrict__ wp2, const float* __restrict__ b2,
    const float* __restrict__ g2, const float* __restrict__ be2,
    const _Float16* __restrict__ wp3, const float* __restrict__ b3,
    float* __restrict__ pool_sum, float* __restrict__ pool_max,
    float* __restrict__ ce_out, int mode) {
  __shared__ _Float16 sA[2][ROWS][APITCH];
  const int lane  = threadIdx.x & 31;
  const int wave  = threadIdx.x >> 5;
  const int strip = wave * 16;
  const int lo16  = lane & 15;
  const int hi    = (lane >= 16) ? 1 : 0;
  const int rowbase = blockIdx.x * ROWS;

  if (!gather) {  // prefetch this WG's contiguous input tile (64 rows x 512B)
    const char* pb = (const char*)(x + (size_t)rowbase * HID);
    __builtin_prefetch(pb + threadIdx.x * 256, 0, 0);
    __builtin_prefetch(pb + threadIdx.x * 256 + 128, 0, 0);
  }

  // ----- Layer 1: x @ W1 + b1 -> LN -> GELU -> sA[0]
  const int arow_g = rowbase + strip + lo16;
  size_t srow;
  if (gather) {
    int b0 = arow_g >> 6;  // 64 gathered rows per batch (K*2)
    srow = (size_t)b0 * NN + gather[arow_g];
  } else {
    srow = (size_t)arow_g;
  }

  v16h af[4];
#pragma unroll
  for (int kt = 0; kt < 4; ++kt)
    af[kt] = a_frag_f32(x + srow * HID, kt * 32 + hi * 8);

  {
    v8f acc[8];
#pragma unroll
    for (int nt = 0; nt < 8; ++nt) {
      float bv = b1[nt * 16 + lo16];
      v8f c;
#pragma unroll
      for (int i = 0; i < 8; ++i) c[i] = bv;
#pragma unroll
      for (int kt = 0; kt < 4; ++kt)
        c = wmma16(af[kt], b_frag(wp1, nt, kt, 4, lane), c);
      acc[nt] = c;
    }
    float gl[8], bl[8];
#pragma unroll
    for (int nt = 0; nt < 8; ++nt) {
      int col = nt * 16 + lo16;
      gl[nt] = g1[col];
      bl[nt] = be1[col];
    }
#pragma unroll
    for (int r = 0; r < 8; ++r) {
      float s = 0.0f, q = 0.0f;
#pragma unroll
      for (int nt = 0; nt < 8; ++nt) {
        float v = acc[nt][r];
        s += v;
        q += v * v;
      }
#pragma unroll
      for (int m = 1; m < 16; m <<= 1) {
        s += __shfl_xor(s, m, 32);
        q += __shfl_xor(q, m, 32);
      }
      float mean = s * (1.0f / 128.0f);
      float inv  = rsqrtf(q * (1.0f / 128.0f) - mean * mean + 1e-5f);
      int row = strip + r + hi * 8;
#pragma unroll
      for (int nt = 0; nt < 8; ++nt) {
        float v = (acc[nt][r] - mean) * inv * gl[nt] + bl[nt];
        sA[0][row][nt * 16 + lo16] = (_Float16)geluf(v);
      }
    }
  }
  __syncthreads();

  // ----- Layer 2: sA[0] @ W2 + b2 -> LN -> GELU -> sA[1]
  {
    v16h a2[4];
#pragma unroll
    for (int kt = 0; kt < 4; ++kt)
      a2[kt] = a_frag_lds(&sA[0][strip + lo16][0], kt * 32 + hi * 8);
    v8f acc[8];
#pragma unroll
    for (int nt = 0; nt < 8; ++nt) {
      float bv = b2[nt * 16 + lo16];
      v8f c;
#pragma unroll
      for (int i = 0; i < 8; ++i) c[i] = bv;
#pragma unroll
      for (int kt = 0; kt < 4; ++kt)
        c = wmma16(a2[kt], b_frag(wp2, nt, kt, 4, lane), c);
      acc[nt] = c;
    }
    float gl[8], bl[8];
#pragma unroll
    for (int nt = 0; nt < 8; ++nt) {
      int col = nt * 16 + lo16;
      gl[nt] = g2[col];
      bl[nt] = be2[col];
    }
#pragma unroll
    for (int r = 0; r < 8; ++r) {
      float s = 0.0f, q = 0.0f;
#pragma unroll
      for (int nt = 0; nt < 8; ++nt) {
        float v = acc[nt][r];
        s += v;
        q += v * v;
      }
#pragma unroll
      for (int m = 1; m < 16; m <<= 1) {
        s += __shfl_xor(s, m, 32);
        q += __shfl_xor(q, m, 32);
      }
      float mean = s * (1.0f / 128.0f);
      float inv  = rsqrtf(q * (1.0f / 128.0f) - mean * mean + 1e-5f);
      int row = strip + r + hi * 8;
#pragma unroll
      for (int nt = 0; nt < 8; ++nt) {
        float v = (acc[nt][r] - mean) * inv * gl[nt] + bl[nt];
        sA[1][row][nt * 16 + lo16] = (_Float16)geluf(v);
      }
    }
  }
  __syncthreads();

  // ----- Layer 3: sA[1] @ W3 + b3 -> ce ; then pool or store
  {
    v16h a3[4];
#pragma unroll
    for (int kt = 0; kt < 4; ++kt)
      a3[kt] = a_frag_lds(&sA[1][strip + lo16][0], kt * 32 + hi * 8);
    v8f acc[4];
#pragma unroll
    for (int nt = 0; nt < 4; ++nt) {
      float bv = b3[nt * 16 + lo16];
      v8f c;
#pragma unroll
      for (int i = 0; i < 8; ++i) c[i] = bv;
#pragma unroll
      for (int kt = 0; kt < 4; ++kt)
        c = wmma16(a3[kt], b_frag(wp3, nt, kt, 4, lane), c);
      acc[nt] = c;
    }

    if (mode == 0) {
      float mf[8];
#pragma unroll
      for (int r = 0; r < 8; ++r) {
        int gr = rowbase + strip + r + hi * 8;
        mf[r] = mask[gr] ? 1.0f : 0.0f;
      }
      int bb = rowbase >> 12;  // 4096 rows per batch, 64 rows per WG
#pragma unroll
      for (int nt = 0; nt < 4; ++nt) {
        float s = 0.0f, mx = -3.4028235e38f;
#pragma unroll
        for (int r = 0; r < 8; ++r) {
          float v = acc[nt][r] * mf[r];
          s += v;
          mx = fmaxf(mx, v);
        }
        s += __shfl_xor(s, 16, 32);
        mx = fmaxf(mx, __shfl_xor(mx, 16, 32));
        if (lane < 16) {
          atomicAdd(&pool_sum[bb * EMB + nt * 16 + lane], s);
          atomicMaxF(&pool_max[bb * EMB + nt * 16 + lane], mx);
        }
      }
    } else {
#pragma unroll
      for (int r = 0; r < 8; ++r) {
        int g  = rowbase + strip + r + hi * 8;
        int bb = g >> 6;
        int n  = gather[g];
        float mf = mask[(size_t)bb * NN + n] ? 1.0f : 0.0f;
#pragma unroll
        for (int nt = 0; nt < 4; ++nt)
          ce_out[(size_t)g * EMB + nt * 16 + lo16] = acc[nt][r] * mf;
      }
    }
  }
}

// Pair MLP: P[4096 x 128] (= [emb_i | emb_j] rows) -> GELU(P@W1+b1)@W2+b2 -> pe[4096 x 32]
__global__ __launch_bounds__(128) void pair_kernel(
    const float* __restrict__ P, const _Float16* __restrict__ wpe1,
    const float* __restrict__ pb1, const _Float16* __restrict__ wpe2,
    const float* __restrict__ pb2, float* __restrict__ pe_out) {
  __shared__ _Float16 sH[ROWS][PPITCH];
  const int lane  = threadIdx.x & 31;
  const int wave  = threadIdx.x >> 5;
  const int strip = wave * 16;
  const int lo16  = lane & 15;
  const int hi    = (lane >= 16) ? 1 : 0;
  const int rowbase = blockIdx.x * ROWS;

  v16h af[4];
#pragma unroll
  for (int kt = 0; kt < 4; ++kt)
    af[kt] = a_frag_f32(P + (size_t)(rowbase + strip + lo16) * 128, kt * 32 + hi * 8);

  // Layer 1: 128 -> 64, GELU
#pragma unroll
  for (int nt = 0; nt < 4; ++nt) {
    float bv = pb1[nt * 16 + lo16];
    v8f c;
#pragma unroll
    for (int i = 0; i < 8; ++i) c[i] = bv;
#pragma unroll
    for (int kt = 0; kt < 4; ++kt)
      c = wmma16(af[kt], b_frag(wpe1, nt, kt, 4, lane), c);
#pragma unroll
    for (int r = 0; r < 8; ++r)
      sH[strip + r + hi * 8][nt * 16 + lo16] = (_Float16)geluf(c[r]);
  }
  __syncthreads();

  // Layer 2: 64 -> 32
  v16h ag[2];
#pragma unroll
  for (int kt = 0; kt < 2; ++kt)
    ag[kt] = a_frag_lds(&sH[strip + lo16][0], kt * 32 + hi * 8);
#pragma unroll
  for (int nt = 0; nt < 2; ++nt) {
    float bv = pb2[nt * 16 + lo16];
    v8f c;
#pragma unroll
    for (int i = 0; i < 8; ++i) c[i] = bv;
#pragma unroll
    for (int kt = 0; kt < 2; ++kt)
      c = wmma16(ag[kt], b_frag(wpe2, nt, kt, 2, lane), c);
#pragma unroll
    for (int r = 0; r < 8; ++r)
      pe_out[(size_t)(rowbase + strip + r + hi * 8) * 32 + nt * 16 + lo16] = c[r];
  }
}

// Per-batch predictor head.
__global__ __launch_bounds__(128) void pred_kernel(
    const float* __restrict__ pool_sum, const float* __restrict__ pool_max,
    const float* __restrict__ nvals, const float* __restrict__ pe,
    const float* __restrict__ w1, const float* __restrict__ b1,
    const float* __restrict__ g1, const float* __restrict__ be1,
    const float* __restrict__ w2, const float* __restrict__ b2,
    const float* __restrict__ w3, const float* __restrict__ b3,
    float* __restrict__ out) {
  __shared__ float comb[292];
  __shared__ float hb[128], s1[128], s2[64];
  const int b = blockIdx.x;
  const int t = threadIdx.x;
  float n  = nvals[b];
  float nc = fmaxf(n, 1.0f);
  if (t < 64) {
    float ps = pool_sum[b * 64 + t];
    comb[t]       = ps;
    comb[64 + t]  = ps / nc;
    comb[128 + t] = pool_max[b * 64 + t];
  }
  if (t < 32) {
    float s = 0.0f, mx = -3.4028235e38f;
    for (int k = 0; k < KPAIR; ++k) {
      float v = pe[(size_t)(b * KPAIR + k) * 32 + t];
      s += v;
      mx = fmaxf(mx, v);
    }
    comb[192 + t] = s;
    comb[224 + t] = s * (1.0f / 32.0f);
    comb[256 + t] = mx;
  }
  if (t == 0) {
    comb[288] = logf(nc);
    comb[289] = n * (1.0f / 32.0f);
    comb[290] = (n * (1.0f / 32.0f)) * (n * (1.0f / 32.0f));
    comb[291] = 1.0f / nc;
  }
  __syncthreads();

  float h = b1[t];
  for (int i = 0; i < 292; ++i) h += comb[i] * w1[i * 128 + t];
  hb[t] = h;
  __syncthreads();
  float s = 0.0f, q = 0.0f;
  for (int i = 0; i < 128; ++i) {
    float v = hb[i];
    s += v;
    q += v * v;
  }
  float mean = s * (1.0f / 128.0f);
  float inv  = rsqrtf(q * (1.0f / 128.0f) - mean * mean + 1e-5f);
  s1[t] = geluf((h - mean) * inv * g1[t] + be1[t]);
  __syncthreads();
  if (t < 64) {
    float h2 = b2[t];
    for (int i = 0; i < 128; ++i) h2 += s1[i] * w2[i * 64 + t];
    s2[t] = geluf(h2);
  }
  __syncthreads();
  if (t == 0) {
    float o = b3[0];
    for (int i = 0; i < 64; ++i) o += s2[i] * w3[i];
    out[b] = (o > 0.0f) ? (o + log1pf(expf(-o))) : log1pf(expf(o));
  }
}

extern "C" void kernel_launch(void* const* d_in, const int* in_sizes, int n_in,
                              void* d_out, int out_size, void* d_ws, size_t ws_size,
                              hipStream_t stream) {
  (void)in_sizes; (void)n_in; (void)out_size; (void)ws_size;
  const float*         x        = (const float*)d_in[0];
  const unsigned char* mask     = (const unsigned char*)d_in[1];
  const int*           pair_idx = (const int*)d_in[2];
  const float* ce_w1 = (const float*)d_in[3];
  const float* ce_b1 = (const float*)d_in[4];
  const float* ce_g1 = (const float*)d_in[5];
  const float* ce_be1= (const float*)d_in[6];
  const float* ce_w2 = (const float*)d_in[7];
  const float* ce_b2 = (const float*)d_in[8];
  const float* ce_g2 = (const float*)d_in[9];
  const float* ce_be2= (const float*)d_in[10];
  const float* ce_w3 = (const float*)d_in[11];
  const float* ce_b3 = (const float*)d_in[12];
  const float* pe_w1 = (const float*)d_in[13];
  const float* pe_b1 = (const float*)d_in[14];
  const float* pe_w2 = (const float*)d_in[15];
  const float* pe_b2 = (const float*)d_in[16];
  const float* pr_w1 = (const float*)d_in[17];
  const float* pr_b1 = (const float*)d_in[18];
  const float* pr_g1 = (const float*)d_in[19];
  const float* pr_be1= (const float*)d_in[20];
  const float* pr_w2 = (const float*)d_in[21];
  const float* pr_b2 = (const float*)d_in[22];
  const float* pr_w3 = (const float*)d_in[23];
  const float* pr_b3 = (const float*)d_in[24];

  char* ws = (char*)d_ws;
  size_t off = 0;
  auto carve = [&](size_t bytes) {
    char* p = ws + off;
    off += (bytes + 255) & ~(size_t)255;
    return p;
  };
  _Float16* wp1   = (_Float16*)carve(128 * 128 * sizeof(_Float16));
  _Float16* wp2   = (_Float16*)carve(128 * 128 * sizeof(_Float16));
  _Float16* wp3   = (_Float16*)carve(128 * 64 * sizeof(_Float16));
  _Float16* wpe1  = (_Float16*)carve(128 * 64 * sizeof(_Float16));
  _Float16* wpe2  = (_Float16*)carve(64 * 32 * sizeof(_Float16));
  float* pool_sum = (float*)carve(BB * EMB * sizeof(float));
  float* pool_max = (float*)carve(BB * EMB * sizeof(float));
  float* nvals    = (float*)carve(BB * sizeof(float));
  float* ce_pairs = (float*)carve((size_t)BB * KPAIR * 2 * EMB * sizeof(float)); // 8192 x 64
  float* pe_out   = (float*)carve((size_t)BB * KPAIR * 32 * sizeof(float));      // 4096 x 32

  // Pack all GEMM weights into WMMA B-fragment layout (f16).
  pack_kernel<<<(128 * 128 + 255) / 256, 256, 0, stream>>>(ce_w1, wp1, 128, 128);
  pack_kernel<<<(128 * 128 + 255) / 256, 256, 0, stream>>>(ce_w2, wp2, 128, 128);
  pack_kernel<<<(128 * 64 + 255) / 256, 256, 0, stream>>>(ce_w3, wp3, 128, 64);
  pack_kernel<<<(128 * 64 + 255) / 256, 256, 0, stream>>>(pe_w1, wpe1, 128, 64);
  pack_kernel<<<(64 * 32 + 255) / 256, 256, 0, stream>>>(pe_w2, wpe2, 64, 32);

  init_kernel<<<(BB * EMB + 255) / 256, 256, 0, stream>>>(pool_sum, pool_max);
  nvals_kernel<<<BB, 256, 0, stream>>>(mask, nvals);

  // Main encoder + masked pooling (524288 rows, 64 rows / WG).
  ce_kernel<<<(BB * NN) / ROWS, 128, 0, stream>>>(
      x, mask, nullptr, wp1, ce_b1, ce_g1, ce_be1, wp2, ce_b2, ce_g2, ce_be2,
      wp3, ce_b3, pool_sum, pool_max, nullptr, 0);

  // Encoder re-run over the 8192 gathered pair rows -> ce_pairs.
  ce_kernel<<<(BB * KPAIR * 2) / ROWS, 128, 0, stream>>>(
      x, mask, pair_idx, wp1, ce_b1, ce_g1, ce_be1, wp2, ce_b2, ce_g2, ce_be2,
      wp3, ce_b3, nullptr, nullptr, ce_pairs, 1);

  // Pair MLP over [4096 x 128] (concat(emb_i, emb_j) = two consecutive rows).
  pair_kernel<<<(BB * KPAIR) / ROWS, 128, 0, stream>>>(ce_pairs, wpe1, pe_b1,
                                                       wpe2, pe_b2, pe_out);

  // Predictor head, one WG per batch element.
  pred_kernel<<<BB, 128, 0, stream>>>(pool_sum, pool_max, nvals, pe_out, pr_w1,
                                      pr_b1, pr_g1, pr_be1, pr_w2, pr_b2, pr_w3,
                                      pr_b3, (float*)d_out);
}
